// Rank_loss_9543417332489
// MI455X (gfx1250) — compile-verified, matrix operand order checked by reading, not run on hardware
//
#include <hip/hip_runtime.h>
#include <hip/hip_bf16.h>
#include <math.h>

typedef __bf16 bf16_t;
typedef bf16_t v16bf __attribute__((ext_vector_type(16)));
typedef bf16_t v8bf  __attribute__((ext_vector_type(8)));
typedef float  v8f   __attribute__((ext_vector_type(8)));

#define MARGIN_1 1.0f
#define MARGIN_2 1.5f
#define ALPHA_1  2.4f
#define ALPHA_2  2.2f
#define TVAL     1.0f
#define EPSV     1e-5f

struct Frag { v8bf lo, hi; };

static __device__ __forceinline__ Frag load_frag(const bf16_t* row, int c0, int c1) {
    Frag f;
    f.lo = *(const v8bf*)(row + c0);
    f.hi = *(const v8bf*)(row + c1);
    return f;
}

static __device__ __forceinline__ v16bf frag_cat(const Frag& f) {
    v16bf a;
    #pragma unroll
    for (int e = 0; e < 8; ++e) { a[e] = f.lo[e]; a[e + 8] = f.hi[e]; }
    return a;
}

// ---------------------------------------------------------------------------
// Pass 0: zero the per-row accumulator array
// ---------------------------------------------------------------------------
__global__ void rl_zero(float* __restrict__ p, int n) {
    int i = blockIdx.x * blockDim.x + threadIdx.x;
    if (i < n) p[i] = 0.0f;
}

// ---------------------------------------------------------------------------
// Pass 1: L2-normalize rows, emit bf16 matrix + fp32 sq[i].
// One 256-thread block (8 wave32) per row.
// ---------------------------------------------------------------------------
__global__ void __launch_bounds__(256)
rl_normalize(const float* __restrict__ x,
             __hip_bfloat16* __restrict__ xn,
             float* __restrict__ sq, int D) {
    const int row = blockIdx.x;
    const float* xr = x + (size_t)row * D;
    __hip_bfloat16* xo = xn + (size_t)row * D;

    __shared__ float wsum[8];
    const int wave = threadIdx.x >> 5;
    const int lane = threadIdx.x & 31;

    float s = 0.0f;
    for (int c = threadIdx.x; c < D; c += blockDim.x) {
        float v = xr[c];
        s += v * v;
    }
    #pragma unroll
    for (int m = 16; m >= 1; m >>= 1) s += __shfl_xor(s, m, 32);
    if (lane == 0) wsum[wave] = s;
    __syncthreads();
    float tot = 0.0f;
    #pragma unroll
    for (int w = 0; w < 8; ++w) tot += wsum[w];
    const float inv = 1.0f / fmaxf(sqrtf(tot), 1e-12f);

    float s2 = 0.0f;
    for (int c = threadIdx.x; c < D; c += blockDim.x) {
        float vn = xr[c] * inv;
        xo[c] = __float2bfloat16(vn);
        s2 += vn * vn;
    }
    #pragma unroll
    for (int m = 16; m >= 1; m >>= 1) s2 += __shfl_xor(s2, m, 32);
    __syncthreads();                 // all reads of wsum done before overwrite
    if (lane == 0) wsum[wave] = s2;
    __syncthreads();
    if (threadIdx.x == 0) {
        float t2 = 0.0f;
        #pragma unroll
        for (int w = 0; w < 8; ++w) t2 += wsum[w];
        sq[row] = t2;
    }
}

// ---------------------------------------------------------------------------
// Pass 2: Gram tile via v_wmma_f32_16x16x32_bf16 + fused loss epilogue.
// Block = 8 waves; wave w owns rows [Mb+32w, Mb+32w+32) x cols [Nb, Nb+64)
// (2 A-fragments x 4 B-fragments = 8 WMMAs per k-step, ~21 FLOP/B).
// Fragments are register double-buffered: loads for k+32 are issued before
// the WMMAs for k so VMEM latency overlaps the XDL pipeline.
// Gram trick: the bf16 B-fragment (32x16, row-striped across lanes) of column
// tile j0 equals the A-fragment of row tile j0 of the same row-major matrix,
// so A and B use identical per-lane addressing (no transpose needed).
// ---------------------------------------------------------------------------
__global__ void __launch_bounds__(256)
rl_gram_epilogue(const __hip_bfloat16* __restrict__ xn,
                 const float* __restrict__ sq,
                 const int* __restrict__ targets,
                 const int* __restrict__ sub,
                 float* __restrict__ acc, int D) {
    const int Nb   = blockIdx.x * 64;    // column tile base
    const int Mb   = blockIdx.y * 256;   // row super-tile base
    const int wave = threadIdx.x >> 5;
    const int lane = threadIdx.x & 31;
    const int g    = lane >> 4;          // lane group (K-half selector)
    const int h    = lane & 15;          // row-within-tile (A) / col (B,C)

    __shared__ int   tj_s[64], sj_s[64];
    __shared__ float sqj_s[64];
    __shared__ int   ti_s[256], si_s[256];
    __shared__ float sqi_s[256];
    if (threadIdx.x < 64) {
        const int j = Nb + threadIdx.x;
        tj_s[threadIdx.x]  = targets[j];
        sj_s[threadIdx.x]  = sub[j];
        sqj_s[threadIdx.x] = sq[j];
    }
    {
        const int i = Mb + threadIdx.x;
        ti_s[threadIdx.x]  = targets[i];
        si_s[threadIdx.x]  = sub[i];
        sqi_s[threadIdx.x] = sq[i];
    }
    __syncthreads();

    const bf16_t* X = reinterpret_cast<const bf16_t*>(xn);
    const size_t sD = (size_t)D;
    const bf16_t* arow[2];
    #pragma unroll
    for (int u = 0; u < 2; ++u)
        arow[u] = X + (size_t)(Mb + wave * 32 + u * 16 + h) * sD;
    const bf16_t* brow[4];
    #pragma unroll
    for (int t = 0; t < 4; ++t)
        brow[t] = X + (size_t)(Nb + t * 16 + h) * sD;

    v8f cacc[2][4];
    #pragma unroll
    for (int u = 0; u < 2; ++u)
        #pragma unroll
        for (int t = 0; t < 4; ++t) cacc[u][t] = v8f{};

    // double-buffered fragments (parity-indexed; unroll 2 makes indices const)
    Frag fa[2][2], fb[2][4];
    {
        const int c0 = 8 * g, c1 = 16 + 8 * g;
        #pragma unroll
        for (int u = 0; u < 2; ++u) fa[0][u] = load_frag(arow[u], c0, c1);
        #pragma unroll
        for (int t = 0; t < 4; ++t) fb[0][t] = load_frag(brow[t], c0, c1);
    }

    #pragma unroll 2
    for (int k0 = 0; k0 < D; k0 += 32) {
        const int p  = (k0 >> 5) & 1;
        const int kn = k0 + 32;
        if (kn < D) {                       // uniform (scalar) branch
            const int c0 = kn + 8 * g, c1 = kn + 16 + 8 * g;
            #pragma unroll
            for (int u = 0; u < 2; ++u) fa[p ^ 1][u] = load_frag(arow[u], c0, c1);
            #pragma unroll
            for (int t = 0; t < 4; ++t) fb[p ^ 1][t] = load_frag(brow[t], c0, c1);
        }
        const v16bf A0 = frag_cat(fa[p][0]);
        const v16bf A1 = frag_cat(fa[p][1]);
        #pragma unroll
        for (int t = 0; t < 4; ++t) {
            const v16bf B = frag_cat(fb[p][t]);
            cacc[0][t] = __builtin_amdgcn_wmma_f32_16x16x32_bf16(
                false, A0, false, B, (short)0, cacc[0][t], false, false);
            cacc[1][t] = __builtin_amdgcn_wmma_f32_16x16x32_bf16(
                false, A1, false, B, (short)0, cacc[1][t], false, false);
        }
    }

    // ---- fused loss epilogue ----
    // C layout: lane (g,h) VGPR r holds D[M = r + 8g][N = h] of each 16x16 tile.
    #pragma unroll
    for (int u = 0; u < 2; ++u) {
        #pragma unroll
        for (int r = 0; r < 8; ++r) {
            const int il = wave * 32 + u * 16 + r + 8 * g;  // row within block
            const int i  = Mb + il;                         // global row
            const int   ti  = ti_s[il];
            const int   si  = si_s[il];
            const float sqi = sqi_s[il];

            float q[8] = {0.f, 0.f, 0.f, 0.f, 0.f, 0.f, 0.f, 0.f};
            #pragma unroll
            for (int t = 0; t < 4; ++t) {
                const int jl = t * 16 + h;
                const int j  = Nb + jl;
                const float dot  = cacc[u][t][r];
                const float d2   = sqi + sqj_s[jl] - 2.0f * dot;
                const float dist = sqrtf(fmaxf(d2, 1e-12f));
                const bool same  = (ti == tj_s[jl]);
                const bool diag  = (i == j);
                const bool intra = (si == sj_s[jl]);
                if (same && !diag) {                       // positive pair
                    if (intra) { q[0] += fmaxf(dist + (MARGIN_1 - ALPHA_1), 0.f); q[1] += 1.f; }
                    else       { q[2] += fmaxf(dist + (MARGIN_2 - ALPHA_2), 0.f); q[3] += 1.f; }
                } else if (!same) {                        // negative pair
                    if (intra) {
                        if (dist < ALPHA_1) {
                            float diff = ALPHA_1 - dist;
                            float w = expf(TVAL * diff);
                            q[4] += diff * w; q[5] += w;
                        }
                    } else {
                        if (dist < ALPHA_2) {
                            float diff = ALPHA_2 - dist;
                            float w = expf(TVAL * diff);
                            q[6] += diff * w; q[7] += w;
                        }
                    }
                }
            }
            // reduce each quantity across the 16 lanes of this group, then one
            // atomic per (row, quantity) from lane h==0.
            #pragma unroll
            for (int qi = 0; qi < 8; ++qi) {
                float v = q[qi];
                v += __shfl_xor(v, 1, 32);
                v += __shfl_xor(v, 2, 32);
                v += __shfl_xor(v, 4, 32);
                v += __shfl_xor(v, 8, 32);
                if (h == 0) atomicAdd(&acc[(size_t)i * 8 + qi], v);
            }
        }
    }
}

// ---------------------------------------------------------------------------
// Pass 3: fold per-row accumulators into row losses, reduce, write sum/N.
// Single 256-thread block.
// ---------------------------------------------------------------------------
__global__ void __launch_bounds__(256)
rl_finalize(const float* __restrict__ acc, float* __restrict__ out, int N) {
    __shared__ float red[256];
    float s = 0.0f;
    for (int i = threadIdx.x; i < N; i += 256) {
        const float* a = acc + (size_t)i * 8;
        s += a[0] / (a[1] + EPSV)
           + a[2] / (a[3] + EPSV)
           + a[4] / (a[5] + EPSV)
           + a[6] / (a[7] + EPSV);
    }
    red[threadIdx.x] = s;
    __syncthreads();
    for (int off = 128; off >= 1; off >>= 1) {
        if (threadIdx.x < off) red[threadIdx.x] += red[threadIdx.x + off];
        __syncthreads();
    }
    if (threadIdx.x == 0) out[0] = red[0] / (float)N;
}

// ---------------------------------------------------------------------------
extern "C" void kernel_launch(void* const* d_in, const int* in_sizes, int n_in,
                              void* d_out, int out_size, void* d_ws, size_t ws_size,
                              hipStream_t stream) {
    const float* x       = (const float*)d_in[0];
    const int*   targets = (const int*)d_in[1];
    const int*   sub     = (const int*)d_in[2];

    const int N = in_sizes[1];          // 4096
    const int D = in_sizes[0] / N;      // 2048

    // workspace layout: [ xn bf16 N*D | sq f32 N | acc f32 N*8 ]
    char* ws = (char*)d_ws;
    __hip_bfloat16* xn = (__hip_bfloat16*)ws;
    size_t off = ((size_t)N * (size_t)D * sizeof(__hip_bfloat16) + 255) & ~(size_t)255;
    float* sq = (float*)(ws + off);
    off = (off + (size_t)N * sizeof(float) + 255) & ~(size_t)255;
    float* acc = (float*)(ws + off);

    const int accN = N * 8;
    rl_zero<<<(accN + 255) / 256, 256, 0, stream>>>(acc, accN);
    rl_normalize<<<N, 256, 0, stream>>>(x, xn, sq, D);

    dim3 grid(N / 64, N / 256);         // 64 x 16 = 1024 workgroups
    rl_gram_epilogue<<<grid, 256, 0, stream>>>(xn, sq, targets, sub, acc, D);

    rl_finalize<<<1, 256, 0, stream>>>(acc, (float*)d_out, N);
}
